// MambaLayer_72567767433626
// MI455X (gfx1250) — matrix operand order repaired and checked by reading, not compiled, not run on hardware
//
#include <hip/hip_runtime.h>
#include <stdint.h>

// ---------------- problem constants ----------------
#define B_SZ    2
#define L_SZ    1024
#define DMODEL  1024
#define DINNER  2048
#define DSTATE  16
#define DCONV   4
#define DTRANK  64
#define MROWS   (B_SZ * L_SZ)          // 2048 GEMM rows
#define XDBL_W  (DTRANK + 2 * DSTATE)  // 96

typedef __attribute__((ext_vector_type(16))) __bf16 bf16x16;
typedef __attribute__((ext_vector_type(8)))  float  v8f;

// ---------------- fp32 -> bf16 (RNE) ----------------
__device__ __forceinline__ uint16_t f2bf(float f) {
    uint32_t u = __builtin_bit_cast(uint32_t, f);
    u += 0x7FFFu + ((u >> 16) & 1u);
    return (uint16_t)(u >> 16);
}

__global__ __launch_bounds__(256) void cvt_bf16(const float* __restrict__ in,
                                                uint16_t* __restrict__ out, int n) {
    int i = blockIdx.x * 256 + threadIdx.x;
    if (i < n) out[i] = f2bf(in[i]);
}

// first DTRANK columns of the 96-wide x_dbl matrix -> contiguous bf16 [M,64]
__global__ __launch_bounds__(256) void cvt_dtlow(const float* __restrict__ xdbl,
                                                 uint16_t* __restrict__ out, int n) {
    int i = blockIdx.x * 256 + threadIdx.x;   // n = MROWS*DTRANK
    if (i < n) {
        int m = i >> 6, c = i & 63;
        out[i] = f2bf(xdbl[(size_t)m * XDBL_W + c]);
    }
}

// ---------------- bf16 WMMA GEMM:  C[M,N] = A[M,K] * W[N,K]^T ----------------
// One wave -> 16x16 M-tile x NT N-tiles.  NO branches: grid sized exactly so
// control flow is fully uniform and EXEC stays all-1s around every WMMA.
// Fragment layouts used here (contiguous k, same order for A and B):
//   A (16x32 bf16): m = lane&15; VGPR v   -> K = half*16 + 2v
//   B (32x16 bf16): n = lane&15; VGPR v   -> K = half*16 + 2v
//   C/D f32:        VGPR r -> (M = r + 8*half, N = lane&15)
template <int NT>
__global__ __launch_bounds__(256) void gemm_bf16_wmma(
    const uint16_t* __restrict__ A, const uint16_t* __restrict__ W,
    float* __restrict__ C, int M, int N, int K, int ng) {
    const int lane = threadIdx.x & 31;
    const int wave = (blockIdx.x * 256 + threadIdx.x) >> 5;
    const int tm   = wave / ng;                 // M tile
    const int tn0  = (wave - tm * ng) * NT;     // first N tile
    const int half = lane >> 4;
    const int l16  = lane & 15;

    const uint16_t* Arow = A + (size_t)(tm * 16 + l16) * K;
    const uint16_t* Wrow[NT];
#pragma unroll
    for (int j = 0; j < NT; ++j)
        Wrow[j] = W + (size_t)((tn0 + j) * 16 + l16) * K;

    v8f acc[NT] = {};
    union Frag { uint32_t u[8]; bf16x16 v; };

    for (int k0 = 0; k0 < K; k0 += 32) {
        Frag fa;
        const int ka = k0 + half * 16;
#pragma unroll
        for (int v = 0; v < 4; ++v) {
            fa.u[v]     = *(const uint32_t*)(Arow + ka + 2 * v);
            fa.u[v + 4] = *(const uint32_t*)(Arow + ka + 8 + 2 * v);
        }
        const int kb = k0 + half * 16;
#pragma unroll
        for (int j = 0; j < NT; ++j) {
            Frag fb;
#pragma unroll
            for (int v = 0; v < 8; ++v)
                fb.u[v] = *(const uint32_t*)(Wrow[j] + kb + 2 * v);
            acc[j] = __builtin_amdgcn_wmma_f32_16x16x32_bf16(
                false, fa.v, false, fb.v, (short)0, acc[j], false, false);
        }
    }
#pragma unroll
    for (int j = 0; j < NT; ++j) {
        const int col = (tn0 + j) * 16 + l16;
#pragma unroll
        for (int r = 0; r < 8; ++r) {
            const int row = tm * 16 + half * 8 + r;
            C[(size_t)row * N + col] = acc[j][r];
        }
    }
}

// ---------------- causal depthwise conv (k=4) + bias + SiLU ----------------
__global__ __launch_bounds__(256) void conv_silu(
    const float* __restrict__ xz, const float* __restrict__ cw,
    const float* __restrict__ cb, float* __restrict__ xs) {
    int i = blockIdx.x * 256 + threadIdx.x;            // B*L*DINNER
    if (i >= B_SZ * L_SZ * DINNER) return;
    int d  = i & (DINNER - 1);
    int bl = i >> 11;                                  // b*L + l
    int l  = bl & (L_SZ - 1);
    float acc = cb[d];
#pragma unroll
    for (int t = 0; t < DCONV; ++t) {
        int ls = l + t - (DCONV - 1);
        if (ls >= 0)
            acc += cw[d * DCONV + t] *
                   xz[(size_t)(bl + t - (DCONV - 1)) * (2 * DINNER) + d];
    }
    xs[i] = acc / (1.f + __expf(-acc));                // SiLU
}

// ---------------- dt = softplus(dt_mat + bias), in place ----------------
__global__ __launch_bounds__(256) void dt_softplus(float* __restrict__ dtm,
                                                   const float* __restrict__ bias,
                                                   int n) {
    int i = blockIdx.x * 256 + threadIdx.x;
    if (i < n) {
        float v = dtm[i] + bias[i & (DINNER - 1)];
        dtm[i] = (v > 20.f) ? v : log1pf(__expf(v));
    }
}

// ---------------- selective scan ----------------
// 4 threads per (b,d) channel, 4 states each.  Per-timestep Bm||Cm (32
// contiguous floats of x_dbl) is staged into LDS with the CDNA5 async
// global->LDS path: 8 lanes of wave 0 each move 16B via
// global_load_async_to_lds_b128, drained with s_wait_asynccnt, then published
// to the workgroup by the barrier.  y reduced over 4 lanes with wave32 shuffles.
__global__ __launch_bounds__(256) void selective_scan(
    const float* __restrict__ dt, const float* __restrict__ xs,
    const float* __restrict__ xdbl, const float* __restrict__ A_log,
    float* __restrict__ ys) {
    const int gid = blockIdx.x * 256 + threadIdx.x;    // 4 * B * DINNER
    const int q  = gid & 3;                            // state group
    const int c  = gid >> 2;                           // b*DINNER + d
    const int b  = c >> 11;
    const int d  = c & (DINNER - 1);
    const int n0 = q * 4;
    float Arow[4], h[4] = {};
#pragma unroll
    for (int n = 0; n < 4; ++n)
        Arow[n] = -__expf(A_log[(size_t)d * DSTATE + n0 + n]);

    __shared__ float sBC[2 * DSTATE];                  // [0:16)=Bm, [16:32)=Cm
    // LDS aperture: low 32 bits of the generic pointer are the LDS byte offset
    const uint32_t ldsBase = (uint32_t)(uintptr_t)(&sBC[0]);

    for (int l = 0; l < L_SZ; ++l) {
        const size_t rl = (size_t)b * L_SZ + l;
        if (threadIdx.x < 8) {                         // 8 lanes x 16B = 128B
            uint32_t ldsAddr = ldsBase + threadIdx.x * 16u;
            uint64_t gAddr   = (uint64_t)(uintptr_t)(xdbl + rl * XDBL_W + DTRANK +
                                                     threadIdx.x * 4);
            asm volatile("global_load_async_to_lds_b128 %0, %1, off"
                         :: "v"(ldsAddr), "v"(gAddr) : "memory");
            asm volatile("s_wait_asynccnt 0x0" ::: "memory");
        }
        __syncthreads();
        const float dtv = dt[rl * DINNER + d];
        const float xv  = xs[rl * DINNER + d];
        float y = 0.f;
#pragma unroll
        for (int n = 0; n < 4; ++n) {
            const float dA = __expf(dtv * Arow[n]);
            h[n] = dA * h[n] + dtv * sBC[n0 + n] * xv;
            y += h[n] * sBC[DSTATE + n0 + n];
        }
        y += __shfl_xor(y, 1, 32);
        y += __shfl_xor(y, 2, 32);
        if (q == 0) ys[rl * DINNER + d] = y;
        __syncthreads();
    }
}

// ---------------- y = (ys + D*xs) * silu(z) ----------------
__global__ __launch_bounds__(256) void gate_kernel(
    const float* __restrict__ ys, const float* __restrict__ xs,
    const float* __restrict__ xz, const float* __restrict__ Dv,
    float* __restrict__ out) {
    int i = blockIdx.x * 256 + threadIdx.x;
    if (i >= B_SZ * L_SZ * DINNER) return;
    int d  = i & (DINNER - 1);
    int bl = i >> 11;
    float z = xz[(size_t)bl * (2 * DINNER) + DINNER + d];
    out[i] = (ys[i] + Dv[d] * xs[i]) * (z / (1.f + __expf(-z)));
}

// ---------------- host launcher ----------------
#define GRID(n) ((unsigned)(((n) + 255) / 256))

extern "C" void kernel_launch(void* const* d_in, const int* in_sizes, int n_in,
                              void* d_out, int out_size, void* d_ws, size_t ws_size,
                              hipStream_t stream) {
    const float* x          = (const float*)d_in[0];
    const float* in_proj_w  = (const float*)d_in[1];
    const float* conv_w     = (const float*)d_in[2];
    const float* conv_b     = (const float*)d_in[3];
    const float* x_proj_w   = (const float*)d_in[4];
    const float* dt_proj_w  = (const float*)d_in[5];
    const float* dt_proj_b  = (const float*)d_in[6];
    const float* A_log      = (const float*)d_in[7];
    const float* Dv         = (const float*)d_in[8];
    const float* out_proj_w = (const float*)d_in[9];
    float* out = (float*)d_out;

    char* ws = (char*)d_ws;
    size_t off = 0;
    auto alloc = [&](size_t bytes) -> void* {
        void* p = ws + off;
        off = (off + bytes + 255) & ~(size_t)255;
        return p;
    };

    uint16_t* xbf    = (uint16_t*)alloc((size_t)MROWS * DMODEL * 2);
    uint16_t* wInBf  = (uint16_t*)alloc((size_t)2 * DINNER * DMODEL * 2);
    float*    xz     = (float*)   alloc((size_t)MROWS * 2 * DINNER * 4);
    float*    xs     = (float*)   alloc((size_t)MROWS * DINNER * 4);
    uint16_t* xsBf   = (uint16_t*)alloc((size_t)MROWS * DINNER * 2);
    uint16_t* wXpBf  = (uint16_t*)alloc((size_t)XDBL_W * DINNER * 2);
    float*    xdbl   = (float*)   alloc((size_t)MROWS * XDBL_W * 4);
    uint16_t* dtLoBf = (uint16_t*)alloc((size_t)MROWS * DTRANK * 2);
    uint16_t* wDtBf  = (uint16_t*)alloc((size_t)DINNER * DTRANK * 2);
    float*    dtm    = (float*)   alloc((size_t)MROWS * DINNER * 4);
    float*    yscan  = (float*)   alloc((size_t)MROWS * DINNER * 4);
    float*    ygate  = (float*)   alloc((size_t)MROWS * DINNER * 4);
    uint16_t* yBf    = (uint16_t*)alloc((size_t)MROWS * DINNER * 2);
    uint16_t* wOutBf = (uint16_t*)alloc((size_t)DMODEL * DINNER * 2);

    const int ntm = MROWS / 16;   // 128 M-tiles everywhere

    // 1) in_proj: xz = x @ in_proj_w^T   [2048,1024] x [4096,1024]^T
    {
        int nA = MROWS * DMODEL, nW = 2 * DINNER * DMODEL;
        cvt_bf16<<<GRID(nA), 256, 0, stream>>>(x, xbf, nA);
        cvt_bf16<<<GRID(nW), 256, 0, stream>>>(in_proj_w, wInBf, nW);
        int ng = (2 * DINNER / 16) / 4;                       // 64
        gemm_bf16_wmma<4><<<ntm * ng / 8, 256, 0, stream>>>(
            xbf, wInBf, xz, MROWS, 2 * DINNER, DMODEL, ng);
    }

    // 2) causal depthwise conv + SiLU -> xs
    conv_silu<<<GRID(MROWS * DINNER), 256, 0, stream>>>(xz, conv_w, conv_b, xs);

    // 3) x_proj: x_dbl = xs @ x_proj_w^T   [2048,2048] x [96,2048]^T
    {
        int nA = MROWS * DINNER, nW = XDBL_W * DINNER;
        cvt_bf16<<<GRID(nA), 256, 0, stream>>>(xs, xsBf, nA);
        cvt_bf16<<<GRID(nW), 256, 0, stream>>>(x_proj_w, wXpBf, nW);
        int ng = (XDBL_W / 16) / 2;                           // 3
        gemm_bf16_wmma<2><<<ntm * ng / 8, 256, 0, stream>>>(
            xsBf, wXpBf, xdbl, MROWS, XDBL_W, DINNER, ng);
    }

    // 4) dt_proj + softplus
    {
        int nL = MROWS * DTRANK, nW = DINNER * DTRANK;
        cvt_dtlow<<<GRID(nL), 256, 0, stream>>>(xdbl, dtLoBf, nL);
        cvt_bf16<<<GRID(nW), 256, 0, stream>>>(dt_proj_w, wDtBf, nW);
        int ng = (DINNER / 16) / 4;                           // 32
        gemm_bf16_wmma<4><<<ntm * ng / 8, 256, 0, stream>>>(
            dtLoBf, wDtBf, dtm, MROWS, DINNER, DTRANK, ng);
        dt_softplus<<<GRID(MROWS * DINNER), 256, 0, stream>>>(
            dtm, dt_proj_b, MROWS * DINNER);
    }

    // 5) selective scan (sequential over L)
    selective_scan<<<GRID(4 * B_SZ * DINNER), 256, 0, stream>>>(
        dtm, xs, xdbl, A_log, yscan);

    // 6) gate: y = (yscan + D*xs) * silu(z)
    gate_kernel<<<GRID(MROWS * DINNER), 256, 0, stream>>>(yscan, xs, xz, Dv, ygate);

    // 7) out_proj -> d_out   [2048,2048] x [1024,2048]^T
    {
        int nA = MROWS * DINNER, nW = DMODEL * DINNER;
        cvt_bf16<<<GRID(nA), 256, 0, stream>>>(ygate, yBf, nA);
        cvt_bf16<<<GRID(nW), 256, 0, stream>>>(out_proj_w, wOutBf, nW);
        int ng = (DMODEL / 16) / 4;                           // 16
        gemm_bf16_wmma<4><<<ntm * ng / 8, 256, 0, stream>>>(
            yBf, wOutBf, out, MROWS, DMODEL, DINNER, ng);
    }
}
